// DMRLModel_33560874451740
// MI455X (gfx1250) — matrix operand order repaired and checked by reading, not compiled, not run on hardware
//
#include <hip/hip_runtime.h>
#include <hip/hip_bf16.h>
#include <stdint.h>

// ---------------- types ----------------
typedef __attribute__((ext_vector_type(16))) _Float16 v16h;
typedef __attribute__((ext_vector_type(8)))  float    v8f;
typedef __attribute__((ext_vector_type(4)))  unsigned int u32x4;
typedef __attribute__((ext_vector_type(4)))  int      i32x4v;
typedef __attribute__((ext_vector_type(8)))  int      i32x8v;

// ---------------- problem constants ----------------
#define BATCH   16384
#define NNEG    5            // N = 1 + NUM_NEG
#define ROWS    (BATCH*NNEG) // 81920
#define TDIM    384
#define HIDP    160          // HID=150 padded to 160
#define DDIM    128
#define FDIM    4
#define FS      32
#define WAVES   8            // waves per block
#define TILES_PER_BLK WAVES
#define NBLOCKS (ROWS/16/TILES_PER_BLK)   // 5120/8 = 640

// fragment-major packed f16 weights: [kt][nt][lane(32)][8 dwords]
// each dword = 2 consecutive-K f16 for one (K-pair, col) per the B-matrix lane layout
#define W1_KT 12             // 384/32
#define W1_NT 10             // 160/16
#define W2_KT 5              // 160/32
#define W2_NT 8              // 128/16
#define W1P_U32 (W1_KT*W1_NT*32*8)   // 30720 u32 = 122880 B
#define W2P_U32 (W2_KT*W2_NT*32*8)   // 10240 u32 = 40960  B

// LDS layout (bytes)
#define LDS_W1   0
#define LDS_W2   122880
#define LDS_HID  163840                  // 8 waves * 16*160 f16 = 40960 B
#define LDS_TE   204800                  // 8 waves * 16*128 f32 = 65536 B
#define LDS_TOTAL 270336

// workspace layout (bytes)
#define WS_W1P    0
#define WS_W2P    122880
#define WS_B1PAD  163840                 // 160 f32
#define WS_LOGITS 164864                 // ROWS*8 f32 = 2,621,440 B
#define WS_SUI    2786304                // ROWS*4 f32 = 1,310,720 B
#define WS_SUT    4097024                // ROWS*4 f32
#define WS_CMAX   5407744                // 40 f32
#define WS_CSUM   5407936                // 40 f32

union FragU {
    v16h     h;
    uint32_t u[8];
    uint4    q[2];
};

static __device__ __forceinline__ uint32_t pack2h(float a, float b) {
    union { _Float16 h[2]; uint32_t u; } x;
    x.h[0] = (_Float16)a; x.h[1] = (_Float16)b;
    return x.u;
}
static __device__ __forceinline__ float leaky(float x) { return x > 0.f ? x : 0.01f * x; }
static __device__ __forceinline__ float softplus(float x) { return x > 20.f ? x : log1pf(expf(x)); }

// ---------------- TDM: 1-D bulk copy global -> LDS (data_size = 8B) ----------------
static __device__ __forceinline__ void tdm_load_1d(uint32_t lds_addr, const void* gptr,
                                                   uint32_t n_qwords) {
    uint64_t ga = (uint64_t)(uintptr_t)gptr;
    u32x4 g0;
    g0[0] = 1u;                                     // count=1, user descriptor
    g0[1] = lds_addr;                               // LDS byte address
    g0[2] = (uint32_t)ga;                           // global_addr[31:0]
    g0[3] = (uint32_t)((ga >> 32) & 0x1FFFFFFu) | (2u << 30);  // addr[56:32] | type=2
    i32x8v g1;
    g1[0] = (int)(3u << 16);                        // workgroup_mask=0, data_size=3 (8B)
    g1[1] = (int)((n_qwords & 0xFFFFu) << 16);      // tensor_dim0[15:0]
    g1[2] = (int)(((n_qwords >> 16) & 0xFFFFu) | (1u << 16)); // dim0 hi | tensor_dim1=1
    g1[3] = (int)((n_qwords & 0xFFFFu) << 16);      // tile_dim0 = n_qwords
    g1[4] = 0;                                      // tile_dim1=0 (unused), tile_dim2=0
    g1[5] = (int)n_qwords;                          // tensor_dim0_stride[31:0]
    g1[6] = 0;
    g1[7] = 0;
    i32x4v z4 = {0, 0, 0, 0};
#if defined(__clang_major__) && __clang_major__ >= 23
    i32x8v z8 = {0, 0, 0, 0, 0, 0, 0, 0};
    __builtin_amdgcn_tensor_load_to_lds(g0, g1, z4, z4, z8, 0);
#else
    __builtin_amdgcn_tensor_load_to_lds(g0, g1, z4, z4, 0);
#endif
}

// ---------------- kernel 0: weight conversion into fragment-major layout ----------------
__global__ void dmrl_prep(const float* __restrict__ w1, const float* __restrict__ b1,
                          const float* __restrict__ w2,
                          uint32_t* __restrict__ w1p, uint32_t* __restrict__ w2p,
                          float* __restrict__ b1pad) {
    int i = blockIdx.x * blockDim.x + threadIdx.x;
    if (i < W1P_U32) {
        // i = (((kt*10 + nt)*32 + L)*8 + j)
        const int j  = i & 7;
        const int L  = (i >> 3) & 31;
        const int q  = i >> 8;             // kt*10 + nt
        const int nt = q % W1_NT;
        const int kt = q / W1_NT;
        const int half = L >> 4;
        const int col  = nt * 16 + (L & 15);
        const int k    = 2 * (kt * 16 + half * 4 + (j & 3) + 8 * (j >> 2));
        float a0 = 0.f, a1 = 0.f;
        if (col < 150) { a0 = w1[col * TDIM + k]; a1 = w1[col * TDIM + k + 1]; }
        w1p[i] = pack2h(a0, a1);
    } else if (i < W1P_U32 + W2P_U32) {
        const int m  = i - W1P_U32;
        const int j  = m & 7;
        const int L  = (m >> 3) & 31;
        const int q  = m >> 8;             // kt*8 + nt
        const int nt = q & 7;
        const int kt = q >> 3;
        const int half = L >> 4;
        const int col  = nt * 16 + (L & 15);      // < 128
        const int k    = 2 * (kt * 16 + half * 4 + (j & 3) + 8 * (j >> 2));  // < 160, pad K>=150 with 0
        const float a0 = (k < 150) ? w2[col * 150 + k] : 0.f;
        const float a1 = (k + 1 < 150) ? w2[col * 150 + k + 1] : 0.f;
        w2p[m] = pack2h(a0, a1);
    } else if (i < W1P_U32 + W2P_U32 + HIDP) {
        const int j = i - W1P_U32 - W2P_U32;
        b1pad[j] = (j < 150) ? b1[j] : 0.f;
    }
}

// ---------------- kernel 1: fused text MLP (WMMA) + attention logits ----------------
__global__ __launch_bounds__(256) void dmrl_main(
    const int* __restrict__ batch, const float* __restrict__ text,
    const float* __restrict__ utab, const float* __restrict__ itab,
    const uint32_t* __restrict__ w1p_g, const uint32_t* __restrict__ w2p_g,
    const float* __restrict__ b1pad, const float* __restrict__ b2,
    const float* __restrict__ aw1, const float* __restrict__ ab1,
    const float* __restrict__ aw2,
    float* __restrict__ logits, float* __restrict__ suiw, float* __restrict__ sutw) {

    extern __shared__ char smem[];
    const uint4* sW1q = (const uint4*)(smem + LDS_W1);   // fragment-major, 2 uint4 per lane-frag
    const uint4* sW2q = (const uint4*)(smem + LDS_W2);
    _Float16*  sHid = (_Float16*)(smem + LDS_HID);
    float*     sTe  = (float*)(smem + LDS_TE);

    const int tid  = threadIdx.x;
    const int wave = tid >> 5;
    const int lane = tid & 31;
    const int lrow = lane & 15;     // row in A layout, column in C layout
    const int wsel = lane >> 4;     // lane-half select (K/M halves)

    // --- stage weights into LDS via Tensor Data Mover (wave 0 only) ---
    if (tid < 32) {
        uint32_t lds0 = (uint32_t)(uintptr_t)(smem);    // aperture low 32 = LDS offset
        tdm_load_1d(lds0 + LDS_W1, (const void*)w1p_g, (W1P_U32 * 4) / 8);
        tdm_load_1d(lds0 + LDS_W2, (const void*)w2p_g, (W2P_U32 * 4) / 8);
        __builtin_amdgcn_s_wait_tensorcnt(0);
    }
    __syncthreads();

    const int tile = blockIdx.x * TILES_PER_BLK + wave;
    const int row0 = tile * 16;
    const int grow = row0 + lrow;

    _Float16* hW  = sHid + wave * 16 * HIDP;
    float*    teW = sTe  + wave * 16 * DDIM;

    // ================= GEMM1: hid = leaky(text @ w1^T + b1) =================
    v8f acc1[10];
#pragma unroll
    for (int i = 0; i < 10; i++) acc1[i] = (v8f){0,0,0,0,0,0,0,0};

    const float* trow = text + (size_t)grow * TDIM;
    for (int kt = 0; kt < W1_KT; kt++) {
        // A fragment: 16x32 f16. lane (lrow, wsel): K = 32*kt + 8*wsel + {0..7}, +16..+23
        const int kb = kt * 32 + wsel * 8;
        const float4* tp = (const float4*)(trow + kb);
        float4 q0 = tp[0], q1 = tp[1], q2 = tp[4], q3 = tp[5];
        v16h a;
        a[0]=(_Float16)q0.x; a[1]=(_Float16)q0.y; a[2]=(_Float16)q0.z; a[3]=(_Float16)q0.w;
        a[4]=(_Float16)q1.x; a[5]=(_Float16)q1.y; a[6]=(_Float16)q1.z; a[7]=(_Float16)q1.w;
        a[8]=(_Float16)q2.x; a[9]=(_Float16)q2.y; a[10]=(_Float16)q2.z; a[11]=(_Float16)q2.w;
        a[12]=(_Float16)q3.x; a[13]=(_Float16)q3.y; a[14]=(_Float16)q3.z; a[15]=(_Float16)q3.w;

        const uint4* fb = sW1q + ((size_t)(kt * W1_NT) * 32 + lane) * 2;
#pragma unroll
        for (int nt = 0; nt < W1_NT; nt++) {
            FragU bf;
            bf.q[0] = fb[nt * 64 + 0];      // 32 lanes * 2 uint4 per nt
            bf.q[1] = fb[nt * 64 + 1];
            acc1[nt] = __builtin_amdgcn_wmma_f32_16x16x32_f16(
                false, a, false, bf.h, (short)0, acc1[nt], false, false);
        }
    }
    // bias + leaky + f16 store to LDS (C layout -> row-major)
#pragma unroll
    for (int nt = 0; nt < W1_NT; nt++) {
        const int col = nt * 16 + lrow;
        const float bias = b1pad[col];
#pragma unroll
        for (int r = 0; r < 8; r++) {
            const int row = wsel ? (8 + r) : r;
            hW[row * HIDP + col] = (_Float16)leaky(acc1[nt][r] + bias);
        }
    }
    asm volatile("s_wait_dscnt 0" ::: "memory");
    __builtin_amdgcn_wave_barrier();

    // ================= GEMM2: te = leaky(hid @ w2^T + b2) =================
    v8f acc2[8];
#pragma unroll
    for (int i = 0; i < 8; i++) acc2[i] = (v8f){0,0,0,0,0,0,0,0};

    for (int kt = 0; kt < W2_KT; kt++) {
        FragU af;
        const int kb = kt * 32 + wsel * 8;
        af.q[0] = *(const uint4*)(hW + lrow * HIDP + kb);
        af.q[1] = *(const uint4*)(hW + lrow * HIDP + kb + 16);

        const uint4* fb = sW2q + ((size_t)(kt * W2_NT) * 32 + lane) * 2;
#pragma unroll
        for (int nt = 0; nt < W2_NT; nt++) {
            FragU bf;
            bf.q[0] = fb[nt * 64 + 0];
            bf.q[1] = fb[nt * 64 + 1];
            acc2[nt] = __builtin_amdgcn_wmma_f32_16x16x32_f16(
                false, af.h, false, bf.h, (short)0, acc2[nt], false, false);
        }
    }
#pragma unroll
    for (int nt = 0; nt < W2_NT; nt++) {
        const int col = nt * 16 + lrow;
        const float bias = b2[col];
#pragma unroll
        for (int r = 0; r < 8; r++) {
            const int row = wsel ? (8 + r) : r;
            teW[row * DDIM + col] = leaky(acc2[nt][r] + bias);
        }
    }
    asm volatile("s_wait_dscnt 0" ::: "memory");
    __builtin_amdgcn_wave_barrier();

    // ================= phase 2: per-(row, feature) attention logits =================
    // 2 lanes per row; each lane covers features {wsel, wsel+2}
    const uint32_t g2 = (uint32_t)(row0 + lrow);
    const uint32_t b  = g2 / NNEG;
    const uint32_t n  = g2 - b * NNEG;
    const int user = batch[b * (NNEG + 1)];
    const int item = batch[b * (NNEG + 1) + 1 + n];
    const float* ubase = utab + (size_t)user * DDIM;
    const float* ibase = itab + (size_t)item * DDIM;
    const float* tbase = teW + lrow * DDIM;

    for (int ff = 0; ff < 2; ff++) {
        const int f = wsel + 2 * ff;
        const float* up  = ubase + f * FS;
        const float* ip  = ibase + f * FS;
        const float* tp2 = tbase + f * FS;
        const float* a10 = aw1 + (f * 2 + 0) * 96;
        const float* a11 = aw1 + (f * 2 + 1) * 96;
        float suu = 0, sii = 0, stt = 0, sui = 0, sut = 0, d0 = 0, d1 = 0;
        for (int d = 0; d < FS; d++) {
            const float uu = up[d], ii = ip[d], tt = tp2[d];
            suu += uu * uu; sii += ii * ii; stt += tt * tt;
            sui += uu * ii; sut += uu * tt;
            d0 += uu * a10[d] + ii * a10[32 + d] + tt * a10[64 + d];
            d1 += uu * a11[d] + ii * a11[32 + d] + tt * a11[64 + d];
        }
        const float inv = 1.f / fmaxf(sqrtf(suu + sii + stt), 1e-12f);
        const float h0 = tanhf(d0 * inv + ab1[f * 2 + 0]);
        const float h1 = tanhf(d1 * inv + ab1[f * 2 + 1]);
        logits[(size_t)g2 * 8 + f * 2 + 0] = h0 * aw2[f * 4 + 0] + h1 * aw2[f * 4 + 1];
        logits[(size_t)g2 * 8 + f * 2 + 1] = h0 * aw2[f * 4 + 2] + h1 * aw2[f * 4 + 3];
        suiw[(size_t)g2 * 4 + f] = sui;
        sutw[(size_t)g2 * 4 + f] = sut;
    }
}

// ---------------- kernel 2: per-column (n,f,o) softmax stats over batch axis ----------------
__global__ void dmrl_colstats(const float* __restrict__ logits,
                              float* __restrict__ cmax, float* __restrict__ csum) {
    __shared__ float red[256];
    const int c = blockIdx.x;          // 40 columns: c = n*8 + f*2 + o, stride 40 over b
    const int t = threadIdx.x;
    float m = -3.4e38f;
    for (int b = t; b < BATCH; b += 256) m = fmaxf(m, logits[(size_t)b * 40 + c]);
    red[t] = m; __syncthreads();
    for (int s = 128; s > 0; s >>= 1) {
        if (t < s) red[t] = fmaxf(red[t], red[t + s]);
        __syncthreads();
    }
    const float cm = red[0]; __syncthreads();
    float sum = 0.f;
    for (int b = t; b < BATCH; b += 256) sum += expf(logits[(size_t)b * 40 + c] - cm);
    red[t] = sum; __syncthreads();
    for (int s = 128; s > 0; s >>= 1) {
        if (t < s) red[t] += red[t + s];
        __syncthreads();
    }
    if (t == 0) { cmax[c] = cm; csum[c] = red[0]; }
}

// ---------------- kernel 3: finalize ratings ----------------
__global__ void dmrl_finalize(const float* __restrict__ logits,
                              const float* __restrict__ suiw, const float* __restrict__ sutw,
                              const float* __restrict__ cmax, const float* __restrict__ csum,
                              float* __restrict__ out) {
    const int g = blockIdx.x * blockDim.x + threadIdx.x;
    if (g >= ROWS) return;
    const int n = g % NNEG;
    float acc = 0.f;
#pragma unroll
    for (int f = 0; f < FDIM; f++) {
        const int c0 = n * 8 + f * 2, c1 = c0 + 1;
        const float a0 = expf(logits[(size_t)g * 8 + f * 2 + 0] - cmax[c0]) / csum[c0];
        const float a1 = expf(logits[(size_t)g * 8 + f * 2 + 1] - cmax[c1]) / csum[c1];
        acc += a0 * softplus(suiw[(size_t)g * 4 + f]) + a1 * softplus(sutw[(size_t)g * 4 + f]);
    }
    out[g] = acc;
}

// ---------------- host launcher ----------------
extern "C" void kernel_launch(void* const* d_in, const int* in_sizes, int n_in,
                              void* d_out, int out_size, void* d_ws, size_t ws_size,
                              hipStream_t stream) {
    const int*   batch = (const int*)d_in[0];
    const float* text  = (const float*)d_in[1];
    const float* utab  = (const float*)d_in[2];
    const float* itab  = (const float*)d_in[3];
    const float* w1    = (const float*)d_in[4];
    const float* b1    = (const float*)d_in[5];
    const float* w2    = (const float*)d_in[6];
    const float* b2    = (const float*)d_in[7];
    const float* aw1   = (const float*)d_in[8];
    const float* ab1   = (const float*)d_in[9];
    const float* aw2   = (const float*)d_in[10];

    char* ws = (char*)d_ws;
    uint32_t* w1p   = (uint32_t*)(ws + WS_W1P);
    uint32_t* w2p   = (uint32_t*)(ws + WS_W2P);
    float*    b1pad = (float*)(ws + WS_B1PAD);
    float*    logit = (float*)(ws + WS_LOGITS);
    float*    suiw  = (float*)(ws + WS_SUI);
    float*    sutw  = (float*)(ws + WS_SUT);
    float*    cmax  = (float*)(ws + WS_CMAX);
    float*    csum  = (float*)(ws + WS_CSUM);

    const int prepN = W1P_U32 + W2P_U32 + HIDP;
    dmrl_prep<<<(prepN + 255) / 256, 256, 0, stream>>>(w1, b1, w2, w1p, w2p, b1pad);

    dmrl_main<<<NBLOCKS, 256, LDS_TOTAL, stream>>>(
        batch, text, utab, itab, w1p, w2p, b1pad, b2, aw1, ab1, aw2,
        logit, suiw, sutw);

    dmrl_colstats<<<40, 256, 0, stream>>>(logit, cmax, csum);

    dmrl_finalize<<<(ROWS + 255) / 256, 256, 0, stream>>>(logit, suiw, sutw, cmax, csum,
                                                          (float*)d_out);
}
